// TGAT_4604204941967
// MI455X (gfx1250) — compile-verified, matrix operand order checked by reading, not run on hardware
//
#include <hip/hip_runtime.h>
#include <hip/hip_bf16.h>

#define NN   100000
#define EE   3200000
#define FIN  172
#define HID  32
#define NT_N (NN/16)   // 6250 node tiles
#define NT_E (EE/16)   // 200000 edge tiles

typedef __attribute__((ext_vector_type(16))) _Float16 v16h;
typedef __attribute__((ext_vector_type(8)))  float    v8f;

// K index for element i of a 16-bit A/B fragment, given lane half (0/1).
__device__ __forceinline__ int kidx(int i, int half) {
  return i + (half << 3) + (i & 8);
}

__device__ __forceinline__ v8f wmma16(v16h a, v16h b, v8f c) {
  return __builtin_amdgcn_wmma_f32_16x16x32_f16(false, a, false, b, (short)0, c,
                                                false, false);
}

// Monotonic float<->uint mapping for atomicMax on floats.
__device__ __forceinline__ unsigned fkey(float f) {
  unsigned b = __float_as_uint(f);
  return (b & 0x80000000u) ? ~b : (b | 0x80000000u);
}
__device__ __forceinline__ float fdekey(unsigned kk) {
  unsigned b = (kk & 0x80000000u) ? (kk & 0x7fffffffu) : ~kk;
  return __uint_as_float(b);
}
#define KEY_NEG_INF 0x007FFFFFu  // fkey(-inf)

// ---------------------------------------------------------------------------
// K0: pre-swizzle weight matrices (K x 32, f32 row-major) into f16 B-fragment
// order: frag[((kc*2 + colTile)*32 + lane)*16 + i]
// ---------------------------------------------------------------------------
__device__ void conv_frag(const float* W, int Kdim, int NC, _Float16* out,
                          int t0, int nt) {
  int total = NC * 1024;  // NC chunks * 2 tiles * 32 lanes * 16 elems
  for (int idx = t0; idx < total; idx += nt) {
    int i  = idx & 15;
    int L  = (idx >> 4) & 31;
    int tt = (idx >> 9) & 1;
    int kc = idx >> 10;
    int k   = kc * 32 + kidx(i, L >> 4);
    int col = tt * 16 + (L & 15);
    out[idx] = (k < Kdim) ? (_Float16)W[k * 32 + col] : (_Float16)0.0f;
  }
}

__global__ void k_convert(const float* Wlin, const float* Wcomb, const float* Wq,
                          const float* Wk, const float* Wv, const float* We,
                          const float* Wsk, _Float16* fLin, _Float16* fComb,
                          _Float16* fQ, _Float16* fK, _Float16* fV,
                          _Float16* fE, _Float16* fS) {
  int t0 = blockIdx.x * blockDim.x + threadIdx.x;
  int nt = gridDim.x * blockDim.x;
  conv_frag(Wlin,  FIN, 6, fLin,  t0, nt);
  conv_frag(Wcomb, 40,  2, fComb, t0, nt);
  conv_frag(Wq,    32,  1, fQ,    t0, nt);
  conv_frag(Wk,    32,  1, fK,    t0, nt);
  conv_frag(Wv,    32,  1, fV,    t0, nt);
  conv_frag(We,    32,  1, fE,    t0, nt);
  conv_frag(Wsk,   32,  1, fS,    t0, nt);
}

// ---------------------------------------------------------------------------
// K1: h1pre = x @ Wlin + blin   (WMMA, K=172 padded to 192)
// ---------------------------------------------------------------------------
__global__ void k_h1pre(const float* __restrict__ x,
                        const _Float16* __restrict__ fLin,
                        const float* __restrict__ blin,
                        float* __restrict__ h1pre) {
  int lane = threadIdx.x & 31, wid = threadIdx.x >> 5;
  int tile = blockIdx.x * 8 + wid;
  if (tile >= NT_N) return;
  int half = lane >> 4, mloc = lane & 15;
  const float* xrow = x + (long)(tile * 16 + mloc) * FIN;
  v8f c0 = {0,0,0,0,0,0,0,0}, c1 = {0,0,0,0,0,0,0,0};
#pragma unroll
  for (int kc = 0; kc < 6; ++kc) {
    v16h a;
#pragma unroll
    for (int i = 0; i < 16; ++i) {
      int k = kc * 32 + kidx(i, half);
      a[i] = (k < FIN) ? (_Float16)xrow[k] : (_Float16)0.0f;
    }
    v16h b0 = *(const v16h*)(fLin + ((kc * 2 + 0) * 32 + lane) * 16);
    v16h b1 = *(const v16h*)(fLin + ((kc * 2 + 1) * 32 + lane) * 16);
    c0 = wmma16(a, b0, c0);
    c1 = wmma16(a, b1, c1);
  }
  float bl0 = blin[mloc], bl1 = blin[16 + mloc];
#pragma unroll
  for (int r = 0; r < 8; ++r) {
    long m = (long)tile * 16 + r + 8 * half;
    h1pre[m * 32 + mloc]      = c0[r] + bl0;
    h1pre[m * 32 + 16 + mloc] = c1[r] + bl1;
  }
}

// ---------------------------------------------------------------------------
// K2: per-node encoder gating -> concat40 (f16, padded to 64)
// ---------------------------------------------------------------------------
__global__ void k_node_enc(const float* __restrict__ h1pre,
                           const float* __restrict__ node_interval,
                           const float* __restrict__ node_degree,
                           const float* __restrict__ Wtf, const float* __restrict__ btf,
                           const float* __restrict__ Wd,  const float* __restrict__ bd,
                           const float* __restrict__ Wenc,const float* __restrict__ benc,
                           const float* __restrict__ Wx,  const float* __restrict__ bx,
                           _Float16* __restrict__ cat) {
  int n = blockIdx.x * blockDim.x + threadIdx.x;
  if (n >= NN) return;
  float h[32];
#pragma unroll
  for (int j = 0; j < 32; ++j) h[j] = h1pre[(long)n * 32 + j];
  float ti = node_interval[n], dg = node_degree[n];
  float tfe[8], dge[8];
#pragma unroll
  for (int j = 0; j < 8; ++j) { tfe[j] = ti * Wtf[j] + btf[j]; dge[j] = dg * Wd[j] + bd[j]; }
  float p0[8], p1[8];
#pragma unroll
  for (int j = 0; j < 8; ++j) {
    float s0 = benc[j], s1 = benc[j];
#pragma unroll
    for (int k = 0; k < 8; ++k) { s0 += tfe[k] * Wenc[k * 8 + j]; s1 += dge[k] * Wenc[k * 8 + j]; }
    p0[j] = tanhf(s0); p1[j] = tanhf(s1);
  }
  float xp[8];
#pragma unroll
  for (int j = 0; j < 8; ++j) {
    float s = bx[j];
#pragma unroll
    for (int k = 0; k < 32; ++k) s += h[k] * Wx[k * 8 + j];
    xp[j] = tanhf(s);
  }
  float s0 = 0.f, s1 = 0.f;
#pragma unroll
  for (int j = 0; j < 8; ++j) { s0 += p0[j] * xp[j]; s1 += p1[j] * xp[j]; }
  float mm = fmaxf(s0, s1);
  float e0 = expf(s0 - mm), e1 = expf(s1 - mm);
  float inv = 1.0f / (e0 + e1);
  float w0 = e0 * inv, w1 = e1 * inv;
  _Float16* out = cat + (long)n * 64;
#pragma unroll
  for (int j = 0; j < 32; ++j) out[j] = (_Float16)h[j];
#pragma unroll
  for (int j = 0; j < 8; ++j) out[32 + j] = (_Float16)(tfe[j] * w0 + dge[j] * w1);
#pragma unroll
  for (int j = 40; j < 64; ++j) out[j] = (_Float16)0.0f;
}

// ---------------------------------------------------------------------------
// K3: h1 = concat40 @ Wcomb + bcomb, then q/k/v/skip = h1 @ W* + b* (WMMA)
// The h1 C-fragment is transposed to A-layout via wave-private LDS.
// ---------------------------------------------------------------------------
__global__ void k_qkvs(const _Float16* __restrict__ cat,
                       const _Float16* __restrict__ fComb, const float* __restrict__ bcomb,
                       const _Float16* __restrict__ fQ, const float* __restrict__ Bq,
                       const _Float16* __restrict__ fK, const float* __restrict__ Bk,
                       const _Float16* __restrict__ fV, const float* __restrict__ Bv,
                       const _Float16* __restrict__ fS, const float* __restrict__ Bs,
                       float* __restrict__ qb, float* __restrict__ kb,
                       float* __restrict__ vb, float* __restrict__ sb) {
  __shared__ _Float16 lds[8][16][32];
  int lane = threadIdx.x & 31, wid = threadIdx.x >> 5;
  int tile = blockIdx.x * 8 + wid;
  bool act = tile < NT_N;
  int half = lane >> 4, mloc = lane & 15;

  if (act) {
    v8f c0 = {0,0,0,0,0,0,0,0}, c1 = {0,0,0,0,0,0,0,0};
    const _Float16* arow = cat + (long)(tile * 16 + mloc) * 64;
#pragma unroll
    for (int kc = 0; kc < 2; ++kc) {
      v16h a;
#pragma unroll
      for (int i = 0; i < 16; ++i) a[i] = arow[kc * 32 + kidx(i, half)];
      v16h b0 = *(const v16h*)(fComb + ((kc * 2 + 0) * 32 + lane) * 16);
      v16h b1 = *(const v16h*)(fComb + ((kc * 2 + 1) * 32 + lane) * 16);
      c0 = wmma16(a, b0, c0);
      c1 = wmma16(a, b1, c1);
    }
    float bc0 = bcomb[mloc], bc1 = bcomb[16 + mloc];
#pragma unroll
    for (int r = 0; r < 8; ++r) {
      int m = r + 8 * half;
      lds[wid][m][mloc]      = (_Float16)(c0[r] + bc0);
      lds[wid][m][16 + mloc] = (_Float16)(c1[r] + bc1);
    }
  }
  __syncthreads();
  if (act) {
    v16h a;
#pragma unroll
    for (int i = 0; i < 16; ++i) a[i] = lds[wid][mloc][kidx(i, half)];
    v16h q0 = *(const v16h*)(fQ + (0 * 32 + lane) * 16);
    v16h q1 = *(const v16h*)(fQ + (1 * 32 + lane) * 16);
    v16h k0 = *(const v16h*)(fK + (0 * 32 + lane) * 16);
    v16h k1 = *(const v16h*)(fK + (1 * 32 + lane) * 16);
    v16h v0 = *(const v16h*)(fV + (0 * 32 + lane) * 16);
    v16h v1 = *(const v16h*)(fV + (1 * 32 + lane) * 16);
    v16h s0 = *(const v16h*)(fS + (0 * 32 + lane) * 16);
    v16h s1 = *(const v16h*)(fS + (1 * 32 + lane) * 16);
    v8f z = {0,0,0,0,0,0,0,0};
    v8f cq0 = wmma16(a, q0, z), cq1 = wmma16(a, q1, z);
    v8f ck0 = wmma16(a, k0, z), ck1 = wmma16(a, k1, z);
    v8f cv0 = wmma16(a, v0, z), cv1 = wmma16(a, v1, z);
    v8f cs0 = wmma16(a, s0, z), cs1 = wmma16(a, s1, z);
    float bq0 = Bq[mloc], bq1 = Bq[16 + mloc];
    float bk0 = Bk[mloc], bk1 = Bk[16 + mloc];
    float bv0 = Bv[mloc], bv1 = Bv[16 + mloc];
    float bs0 = Bs[mloc], bs1 = Bs[16 + mloc];
#pragma unroll
    for (int r = 0; r < 8; ++r) {
      long m = (long)tile * 16 + r + 8 * half;
      qb[m * 32 + mloc] = cq0[r] + bq0;  qb[m * 32 + 16 + mloc] = cq1[r] + bq1;
      kb[m * 32 + mloc] = ck0[r] + bk0;  kb[m * 32 + 16 + mloc] = ck1[r] + bk1;
      vb[m * 32 + mloc] = cv0[r] + bv0;  vb[m * 32 + 16 + mloc] = cv1[r] + bv1;
      sb[m * 32 + mloc] = cs0[r] + bs0;  sb[m * 32 + 16 + mloc] = cs1[r] + bs1;
    }
  }
}

// ---------------------------------------------------------------------------
// K4/K5 shared: build e-tiles for a 16-edge tile fully in registers.
// A fragment = cos(rel_t * Wt + bt) computed per lane; B = fE frags; +be.
// Column tile 0 == head 0 channels, tile 1 == head 1 channels.
// ---------------------------------------------------------------------------
__device__ __forceinline__ void edge_e_tiles(long ebase, int lane,
                                             const int* __restrict__ ei,
                                             const float* __restrict__ node_time,
                                             const float* __restrict__ edge_time,
                                             const float* __restrict__ Wt,
                                             const float* __restrict__ bt,
                                             const _Float16* __restrict__ fE,
                                             const float* __restrict__ be,
                                             v8f& e0, v8f& e1,
                                             int& src_m, int& dst_m) {
  int half = lane >> 4, mloc = lane & 15;
  long my_e = ebase + mloc;
  src_m = ei[my_e];
  dst_m = ei[(long)EE + my_e];
  float rel = node_time[src_m] - edge_time[my_e];
  v16h a;
#pragma unroll
  for (int i = 0; i < 16; ++i) {
    int j = kidx(i, half);
    a[i] = (_Float16)cosf(rel * Wt[j] + bt[j]);
  }
  v16h b0 = *(const v16h*)(fE + (0 * 32 + lane) * 16);
  v16h b1 = *(const v16h*)(fE + (1 * 32 + lane) * 16);
  v8f z = {0,0,0,0,0,0,0,0};
  e0 = wmma16(a, b0, z);
  e1 = wmma16(a, b1, z);
  float be0 = be[mloc], be1 = be[16 + mloc];
#pragma unroll
  for (int r = 0; r < 8; ++r) { e0[r] += be0; e1[r] += be1; }
}

// K4: attention logits + per-dst running max
__global__ void k_edge_alpha(const int* __restrict__ ei,
                             const float* __restrict__ node_time,
                             const float* __restrict__ edge_time,
                             const float* __restrict__ Wt, const float* __restrict__ bt,
                             const _Float16* __restrict__ fE, const float* __restrict__ be,
                             const float* __restrict__ qb, const float* __restrict__ kb,
                             float* __restrict__ alpha_raw, unsigned* __restrict__ amax) {
  int lane = threadIdx.x & 31, wid = threadIdx.x >> 5;
  long tile = (long)blockIdx.x * 8 + wid;
  if (tile >= NT_E) return;
  int half = lane >> 4, mloc = lane & 15;
  long ebase = tile * 16;
  v8f e0, e1; int src_m, dst_m;
  edge_e_tiles(ebase, lane, ei, node_time, edge_time, Wt, bt, fE, be, e0, e1, src_m, dst_m);
#pragma unroll
  for (int r = 0; r < 8; ++r) {
    int lsel = r + (half << 3);
    int s = __shfl(src_m, lsel, 32);
    int d = __shfl(dst_m, lsel, 32);
    long er = ebase + r + 8 * half;
    float p0 = qb[(long)d * 32 + mloc]      * (kb[(long)s * 32 + mloc]      + e0[r]);
    float p1 = qb[(long)d * 32 + 16 + mloc] * (kb[(long)s * 32 + 16 + mloc] + e1[r]);
#pragma unroll
    for (int off = 8; off; off >>= 1) {
      p0 += __shfl_xor(p0, off, 32);
      p1 += __shfl_xor(p1, off, 32);
    }
    p0 *= 0.25f;  // 1/sqrt(16)
    p1 *= 0.25f;
    if (mloc == 0) {
      alpha_raw[er * 2 + 0] = p0;
      alpha_raw[er * 2 + 1] = p1;
      atomicMax(&amax[(long)d * 2 + 0], fkey(p0));
      atomicMax(&amax[(long)d * 2 + 1], fkey(p1));
    }
  }
}

// K5: exp-weights, denominator, and weighted scatter of (v[src]+e)
__global__ void k_edge_agg(const int* __restrict__ ei,
                           const float* __restrict__ node_time,
                           const float* __restrict__ edge_time,
                           const float* __restrict__ Wt, const float* __restrict__ bt,
                           const _Float16* __restrict__ fE, const float* __restrict__ be,
                           const float* __restrict__ vb,
                           const float* __restrict__ alpha_raw,
                           const unsigned* __restrict__ amax,
                           float* __restrict__ asum, float* __restrict__ agg) {
  int lane = threadIdx.x & 31, wid = threadIdx.x >> 5;
  long tile = (long)blockIdx.x * 8 + wid;
  if (tile >= NT_E) return;
  int half = lane >> 4, mloc = lane & 15;
  long ebase = tile * 16;
  v8f e0, e1; int src_m, dst_m;
  edge_e_tiles(ebase, lane, ei, node_time, edge_time, Wt, bt, fE, be, e0, e1, src_m, dst_m);
#pragma unroll
  for (int r = 0; r < 8; ++r) {
    int lsel = r + (half << 3);
    int s = __shfl(src_m, lsel, 32);
    int d = __shfl(dst_m, lsel, 32);
    long er = ebase + r + 8 * half;
    float a0 = expf(alpha_raw[er * 2 + 0] - fdekey(amax[(long)d * 2 + 0]));
    float a1 = expf(alpha_raw[er * 2 + 1] - fdekey(amax[(long)d * 2 + 1]));
    if (mloc == 0) {
      atomicAdd(&asum[(long)d * 2 + 0], a0);
      atomicAdd(&asum[(long)d * 2 + 1], a1);
    }
    float m0 = a0 * (vb[(long)s * 32 + mloc]      + e0[r]);
    float m1 = a1 * (vb[(long)s * 32 + 16 + mloc] + e1[r]);
    atomicAdd(&agg[(long)d * 32 + mloc],      m0);
    atomicAdd(&agg[(long)d * 32 + 16 + mloc], m1);
  }
}

// init scratch accumulators
__global__ void k_init(unsigned* __restrict__ amax, float* __restrict__ asum,
                       float* __restrict__ agg) {
  int t = blockIdx.x * blockDim.x + threadIdx.x;
  if (t < 2 * NN) { amax[t] = KEY_NEG_INF; asum[t] = 0.0f; }
  if (t < NN * HID) agg[t] = 0.0f;
}

// K6: normalize, skip, 32->2, log_softmax
__global__ void k_final(const float* __restrict__ agg, const float* __restrict__ asum,
                        const float* __restrict__ sb, const float* __restrict__ Wout,
                        const float* __restrict__ bout, float* __restrict__ out) {
  int n = blockIdx.x * blockDim.x + threadIdx.x;
  if (n >= NN) return;
  float inv0 = 1.0f / (asum[(long)n * 2 + 0] + 1e-16f);
  float inv1 = 1.0f / (asum[(long)n * 2 + 1] + 1e-16f);
  float o0 = bout[0], o1 = bout[1];
#pragma unroll
  for (int c = 0; c < 32; ++c) {
    float h2 = agg[(long)n * 32 + c] * ((c < 16) ? inv0 : inv1) + sb[(long)n * 32 + c];
    o0 += h2 * Wout[c * 2 + 0];
    o1 += h2 * Wout[c * 2 + 1];
  }
  float mm  = fmaxf(o0, o1);
  float lse = mm + logf(expf(o0 - mm) + expf(o1 - mm));
  out[(long)n * 2 + 0] = o0 - lse;
  out[(long)n * 2 + 1] = o1 - lse;
}

// ---------------------------------------------------------------------------
extern "C" void kernel_launch(void* const* d_in, const int* in_sizes, int n_in,
                              void* d_out, int out_size, void* d_ws, size_t ws_size,
                              hipStream_t stream) {
  (void)in_sizes; (void)n_in; (void)out_size; (void)ws_size;
  const float* x             = (const float*)d_in[0];
  const int*   ei            = (const int*)d_in[1];
  const float* node_time     = (const float*)d_in[2];
  const float* edge_time     = (const float*)d_in[3];
  const float* node_interval = (const float*)d_in[4];
  const float* node_degree   = (const float*)d_in[5];
  const float* Wt   = (const float*)d_in[6];  const float* bt    = (const float*)d_in[7];
  const float* Wd   = (const float*)d_in[8];  const float* bd    = (const float*)d_in[9];
  const float* Wtf  = (const float*)d_in[10]; const float* btf   = (const float*)d_in[11];
  const float* Wenc = (const float*)d_in[12]; const float* benc  = (const float*)d_in[13];
  const float* Wx   = (const float*)d_in[14]; const float* bx    = (const float*)d_in[15];
  const float* Wlin = (const float*)d_in[16]; const float* blin  = (const float*)d_in[17];
  const float* Wcomb= (const float*)d_in[18]; const float* bcomb = (const float*)d_in[19];
  const float* Wq   = (const float*)d_in[20]; const float* bq    = (const float*)d_in[21];
  const float* Wk   = (const float*)d_in[22]; const float* bk    = (const float*)d_in[23];
  const float* Wv   = (const float*)d_in[24]; const float* bv    = (const float*)d_in[25];
  const float* We   = (const float*)d_in[26]; const float* be    = (const float*)d_in[27];
  const float* Wsk  = (const float*)d_in[28]; const float* bsk   = (const float*)d_in[29];
  const float* Wout = (const float*)d_in[30]; const float* bout  = (const float*)d_in[31];

  char* ws = (char*)d_ws;
  size_t off = 0;
  auto alloc = [&](size_t bytes) -> void* {
    void* p = ws + off;
    off = (off + bytes + 255) & ~(size_t)255;
    return p;
  };
  _Float16* fLin  = (_Float16*)alloc(6 * 1024 * 2);
  _Float16* fComb = (_Float16*)alloc(2 * 1024 * 2);
  _Float16* fQ    = (_Float16*)alloc(1024 * 2);
  _Float16* fK    = (_Float16*)alloc(1024 * 2);
  _Float16* fV    = (_Float16*)alloc(1024 * 2);
  _Float16* fE    = (_Float16*)alloc(1024 * 2);
  _Float16* fS    = (_Float16*)alloc(1024 * 2);
  float*    h1pre = (float*)alloc((size_t)NN * 32 * 4);
  _Float16* cat   = (_Float16*)alloc((size_t)NN * 64 * 2);
  float*    qbuf  = (float*)alloc((size_t)NN * 32 * 4);
  float*    kbuf  = (float*)alloc((size_t)NN * 32 * 4);
  float*    vbuf  = (float*)alloc((size_t)NN * 32 * 4);
  float*    sbuf  = (float*)alloc((size_t)NN * 32 * 4);
  float*    alpha = (float*)alloc((size_t)EE * 2 * 4);
  unsigned* amax  = (unsigned*)alloc((size_t)NN * 2 * 4);
  float*    asum  = (float*)alloc((size_t)NN * 2 * 4);
  float*    agg   = (float*)alloc((size_t)NN * 32 * 4);

  k_convert<<<16, 256, 0, stream>>>(Wlin, Wcomb, Wq, Wk, Wv, We, Wsk,
                                    fLin, fComb, fQ, fK, fV, fE, fS);
  k_h1pre<<<(NT_N + 7) / 8, 256, 0, stream>>>(x, fLin, blin, h1pre);
  k_node_enc<<<(NN + 255) / 256, 256, 0, stream>>>(h1pre, node_interval, node_degree,
                                                   Wtf, btf, Wd, bd, Wenc, benc, Wx, bx, cat);
  k_qkvs<<<(NT_N + 7) / 8, 256, 0, stream>>>(cat, fComb, bcomb, fQ, bq, fK, bk,
                                             fV, bv, fS, bsk, qbuf, kbuf, vbuf, sbuf);
  k_init<<<(NN * HID + 255) / 256, 256, 0, stream>>>(amax, asum, agg);
  k_edge_alpha<<<(NT_E + 7) / 8, 256, 0, stream>>>(ei, node_time, edge_time, Wt, bt,
                                                   fE, be, qbuf, kbuf, alpha, amax);
  k_edge_agg<<<(NT_E + 7) / 8, 256, 0, stream>>>(ei, node_time, edge_time, Wt, bt,
                                                 fE, be, vbuf, alpha, amax, asum, agg);
  k_final<<<(NN + 255) / 256, 256, 0, stream>>>(agg, asum, sbuf, Wout, bout,
                                                (float*)d_out);
}
